// VectorQuantizerEMA_15899968930265
// MI455X (gfx1250) — compile-verified, matrix operand order checked by reading, not run on hardware
//
#include <hip/hip_runtime.h>
#include <hip/hip_bf16.h>

// ---------------------------------------------------------------------------
// VQ-VAE EMA vector quantizer for MI455X (gfx1250, wave32, WMMA).
//   z:            (B=32, D=256, H=64, W=64) fp32
//   embedding:    (K=1024, D=256) fp32
//   Outputs (concat, fp32): quantized (33554432) | loss (1) | indices (131072)
//                           | new_cluster_size (1024) | new_ema_embedding (262144)
// Distance GEMM with v_wmma_f32_16x16x32_bf16; codebook B-fragments staged in
// LDS (double-buffered) and shared by 8 waves per workgroup to cut L2 traffic.
// ---------------------------------------------------------------------------

typedef __attribute__((ext_vector_type(16))) __bf16 v16bf;
typedef __attribute__((ext_vector_type(8)))  float  v8f;

#define K_CODES   1024
#define D_DIM     256
#define HW        4096          // H*W
#define N_VECS    131072        // B*H*W
#define Z_ELEMS   33554432      // B*D*H*W
#define NUM_TILES 64            // K_CODES / 16
#define NUM_FRAGS 8             // D_DIM / 32
#define WAVES_PB  8             // waves (16-row tiles) per workgroup
#define DECAY_F   0.99f
#define OMD_F     0.01f
#define BETA_F    0.25f

static __device__ __forceinline__ __bf16 f2bf(float f) {
    // round-to-nearest-even fp32 -> bf16
    unsigned u = __builtin_bit_cast(unsigned, f);
    unsigned r = u + 0x7fffu + ((u >> 16) & 1u);
    unsigned short h = (unsigned short)(r >> 16);
    return __builtin_bit_cast(__bf16, h);
}

// Zero scratch accumulators (dw, counts, loss).
__global__ void vq_zero(float* __restrict__ p, int n) {
    int i = blockIdx.x * blockDim.x + threadIdx.x;
    if (i < n) p[i] = 0.0f;
}

// Pre-swizzle codebook into WMMA bf16 B-fragment layout + compute ||e_k||^2.
// One wave per 16-code tile. ISA 16-bit fragment layout:
//   lane L (<16):  col = L,      vector elem i -> k = 8*half + i + (i>=8 ? 8 : 0)
//   lane L (>=16): col = L-16,   same with half=1
__global__ __launch_bounds__(32)
void vq_prep(const float* __restrict__ emb,
             v16bf* __restrict__ Bp, float* __restrict__ enorm) {
    const int lane = threadIdx.x;
    const int half = lane >> 4;
    const int col  = lane & 15;
    const int t    = blockIdx.x;            // code tile 0..63
    const int code = t * 16 + col;
    const float* er = emb + (size_t)code * D_DIM;

    float sq = 0.0f;
#pragma unroll
    for (int f = 0; f < NUM_FRAGS; ++f) {
        v16bf frag;
#pragma unroll
        for (int i = 0; i < 16; ++i) {
            int k = f * 32 + half * 8 + i + ((i >= 8) ? 8 : 0);
            float v = er[k];
            sq += v * v;
            frag[i] = f2bf(v);
        }
        Bp[(t * NUM_FRAGS + f) * 32 + lane] = frag;  // 32B/lane coalesced
    }
    sq += __shfl_xor(sq, 16, 32);           // halves cover disjoint k
    if (half == 0) enorm[code] = sq;
}

// Main fused kernel: 8 waves per block, each wave a 16-row tile of z_flat.
// Codebook tile (8KB) staged in LDS once per block, double-buffered.
__global__ __launch_bounds__(WAVES_PB * 32)
void vq_main(const float* __restrict__ z, const float* __restrict__ emb,
             const v16bf* __restrict__ Bp, const float* __restrict__ enorm,
             float* __restrict__ counts, float* __restrict__ dw,
             float* __restrict__ loss_acc,
             float* __restrict__ q_out, float* __restrict__ idx_out) {
    const int tid  = threadIdx.x;
    const int wave = tid >> 5;
    const int lane = tid & 31;
    const int half = lane >> 4;
    const int col  = lane & 15;
    const int row_base = (blockIdx.x * WAVES_PB + wave) * 16;
    const int b   = row_base >> 12;         // row / (H*W)
    const int hw0 = row_base & 4095;
    const float* zb = z + (size_t)b * D_DIM * HW;

    // Double-buffered LDS for one 16-code B tile (8 frags x 32 lanes x 32B).
    __shared__ v16bf sB[2][NUM_FRAGS * 32];

    // --- A fragments (16x32 bf16 each) straight from global z, fp32 -> bf16.
    //     For fixed k, the 16 rows are memory-adjacent -> 64B coalesced.
    v16bf A[NUM_FRAGS];
    float zsq = 0.0f;
#pragma unroll
    for (int f = 0; f < NUM_FRAGS; ++f) {
#pragma unroll
        for (int i = 0; i < 16; ++i) {
            int k = f * 32 + half * 8 + i + ((i >= 8) ? 8 : 0);
            float v = zb[(size_t)k * HW + (hw0 + col)];
            zsq += v * v;
            A[f][i] = f2bf(v);
        }
    }

    // Stage tile 0 (each thread copies one 32B fragment entry).
    sB[0][tid] = Bp[tid];
    __syncthreads();

    // --- K loop: 64 code tiles x 8 WMMAs; score = ||e||^2 - 2*(z.e) ---
    v8f minval;
    int minidx[8];
#pragma unroll
    for (int j = 0; j < 8; ++j) { minval[j] = 3.0e38f; minidx[j] = 0; }

    int buf = 0;
    for (int t = 0; t < NUM_TILES; ++t) {
        // Prefetch next tile from L2 into registers (latency hidden by WMMAs).
        v16bf stage;
        if (t + 1 < NUM_TILES) stage = Bp[(size_t)(t + 1) * (NUM_FRAGS * 32) + tid];

        v8f acc = {};
#pragma unroll
        for (int f = 0; f < NUM_FRAGS; ++f) {
            v16bf Bf = sB[buf][f * 32 + lane];   // ds_load 32B/lane
            acc = __builtin_amdgcn_wmma_f32_16x16x32_bf16(
                false, A[f], false, Bf, (short)0, acc, false, false);
        }
        float en = enorm[t * 16 + col];
        int code = t * 16 + col;
#pragma unroll
        for (int j = 0; j < 8; ++j) {
            float s = en - 2.0f * acc[j];
            if (s < minval[j]) { minval[j] = s; minidx[j] = code; }
        }

        if (t + 1 < NUM_TILES) sB[buf ^ 1][tid] = stage;
        __syncthreads();
        buf ^= 1;
    }

    // --- argmin across the 16 columns (lanes within each half-group);
    //     ties -> lowest code index (matches jnp.argmin) ---
#pragma unroll
    for (int off = 1; off < 16; off <<= 1) {
#pragma unroll
        for (int j = 0; j < 8; ++j) {
            float ov = __shfl_xor(minval[j], off, 16);
            int   oi = __shfl_xor(minidx[j], off, 16);
            if (ov < minval[j] || (ov == minval[j] && oi < minidx[j])) {
                minval[j] = ov; minidx[j] = oi;
            }
        }
    }
    zsq += __shfl_xor(zsq, 16, 32);          // full ||z_row||^2, row = col

    __shared__ int   s_idx[WAVES_PB][16];
    __shared__ float s_sc [WAVES_PB][16];
    __shared__ float s_zsq[WAVES_PB][16];
    if (col == 0) {                          // lane 0 -> rows 0..7, lane 16 -> 8..15
#pragma unroll
        for (int j = 0; j < 8; ++j) {
            s_idx[wave][half * 8 + j] = minidx[j];
            s_sc [wave][half * 8 + j] = minval[j];
        }
    }
    if (half == 0) s_zsq[wave][col] = zsq;
    __syncthreads();

    // indices (as float), counts, commitment-loss partial
    if (half == 0) {
        int id = s_idx[wave][col];
        idx_out[row_base + col] = (float)id;
        atomicAdd(&counts[id], 1.0f);
    } else if (lane == 16) {
        float ls = 0.0f;
#pragma unroll
        for (int r = 0; r < 16; ++r) ls += s_zsq[wave][r] + s_sc[wave][r];
        atomicAdd(loss_acc, ls);
    }

    // --- quantized output (gather codebook, coalesced store in z layout)
    //     + dw segment-sum via fp32 atomics (dw is 1MB, L2-resident) ---
    const int   myidx = s_idx[wave][col];
    const float* erow = emb + (size_t)myidx * D_DIM;
    float* qb = q_out + (size_t)b * D_DIM * HW;
    for (int d0 = 0; d0 < D_DIM; d0 += 2) {
        int d = d0 + half;                   // two d's per iteration
        float e = erow[d];
        size_t off = (size_t)d * HW + (hw0 + col);
        qb[off] = e;
        float zv = zb[off];
        atomicAdd(&dw[myidx * D_DIM + d], zv);
    }
}

// EMA blends + loss scaling.
__global__ void vq_final(const float* __restrict__ ema_cs,
                         const float* __restrict__ ema_emb,
                         const float* __restrict__ counts,
                         const float* __restrict__ dw,
                         const float* __restrict__ loss_acc,
                         float* __restrict__ out_cs,
                         float* __restrict__ out_emb,
                         float* __restrict__ out_loss) {
    int i = blockIdx.x * blockDim.x + threadIdx.x;
    if (i < K_CODES * D_DIM) out_emb[i] = DECAY_F * ema_emb[i] + OMD_F * dw[i];
    if (i < K_CODES)         out_cs[i]  = DECAY_F * ema_cs[i]  + OMD_F * counts[i];
    if (i == 0) *out_loss = BETA_F * (*loss_acc) * (1.0f / (float)Z_ELEMS);
}

extern "C" void kernel_launch(void* const* d_in, const int* in_sizes, int n_in,
                              void* d_out, int out_size, void* d_ws, size_t ws_size,
                              hipStream_t stream) {
    (void)in_sizes; (void)n_in; (void)out_size; (void)ws_size;
    const float* z       = (const float*)d_in[0];
    const float* emb     = (const float*)d_in[1];
    const float* ema_cs  = (const float*)d_in[2];
    const float* ema_emb = (const float*)d_in[3];

    // d_out layout (all fp32, reference return order)
    float* q_out    = (float*)d_out;           // 33,554,432
    float* loss_out = q_out + Z_ELEMS;         // 1
    float* idx_out  = loss_out + 1;            // 131,072
    float* cs_out   = idx_out + N_VECS;        // 1,024
    float* emb_out  = cs_out + K_CODES;        // 262,144

    // workspace layout (floats): dw | counts | loss | pad | enorm | Bprep(bf16)
    float* dw     = (float*)d_ws;                       // 262144
    float* counts = dw + K_CODES * D_DIM;               // 1024
    float* loss   = counts + K_CODES;                   // 1 (+7 pad)
    float* enorm  = loss + 8;                           // 1024
    v16bf* Bp     = (v16bf*)(enorm + K_CODES);          // 32B-aligned, 512KB

    const int n_zero = K_CODES * D_DIM + K_CODES + 1;   // dw + counts + loss
    vq_zero <<<(n_zero + 255) / 256, 256, 0, stream>>>((float*)d_ws, n_zero);
    vq_prep <<<NUM_TILES, 32, 0, stream>>>(emb, Bp, enorm);
    vq_main <<<N_VECS / (16 * WAVES_PB), WAVES_PB * 32, 0, stream>>>(
        z, emb, Bp, enorm, counts, dw, loss, q_out, idx_out);
    vq_final<<<(K_CODES * D_DIM + 255) / 256, 256, 0, stream>>>(
        ema_cs, ema_emb, counts, dw, loss, cs_out, emb_out, loss_out);
}
